// AttentionHead_70351564308691
// MI455X (gfx1250) — matrix operand order repaired and checked by reading, not compiled, not run on hardware
//
#include <hip/hip_runtime.h>
#include <hip/hip_bf16.h>

typedef _Float16 v16h __attribute__((ext_vector_type(16)));
typedef _Float16 v8h  __attribute__((ext_vector_type(8)));
typedef _Float16 v4h  __attribute__((ext_vector_type(4)));
typedef float    v8f  __attribute__((ext_vector_type(8)));
typedef unsigned int u32x4 __attribute__((ext_vector_type(4)));
typedef unsigned int u32x8 __attribute__((ext_vector_type(8)));

#define DIM_IN 1024
#define DIM_D  128
#define BATCH  8
#define SEQ    2048
#define MROWS  (BATCH * SEQ)

static __device__ __forceinline__ v8f vzero8() {
    v8f z;
#pragma unroll
    for (int i = 0; i < 8; ++i) z[i] = 0.0f;
    return z;
}

// A-matrix 16x32 f16 fragment for V_WMMA_F32_16X16X32_F16.
// Lane L holds row (L&15); k = hf*8 + {0..7} and 16 + hf*8 + {0..7}, hf = L>>4.
static __device__ __forceinline__ v16h lds_afrag(const _Float16* base, int stride,
                                                 int row, int kcol0, int lane) {
    const int hf = lane >> 4;
    const _Float16* p = base + row * stride + kcol0 + hf * 8;
    v8h lo = *(const v8h*)(p);
    v8h hi = *(const v8h*)(p + 16);
    v16h r;
#pragma unroll
    for (int i = 0; i < 8; ++i) { r[i] = lo[i]; r[i + 8] = hi[i]; }
    return r;
}

// B-matrix 32x16 f16 fragment: lane L holds column n; k = hf*8 + {0..7} and 16 + hf*8 + {0..7}.
// Storage: N-major (K contiguous) f16 tile: base[n*stride + k].
static __device__ __forceinline__ v16h lds_bfrag(const _Float16* base, int stride,
                                                 int ncol, int koff, int lane) {
    const int hf = lane >> 4;
    const _Float16* p = base + ncol * stride + koff + hf * 8;
    v8h lo = *(const v8h*)(p);
    v8h hi = *(const v8h*)(p + 16);
    v16h r;
#pragma unroll
    for (int i = 0; i < 8; ++i) { r[i] = lo[i]; r[i + 8] = hi[i]; }
    return r;
}

// ---------------------------------------------------------------------------
// TDM: async DMA of a [rows x 128] f16 row-major tile (row stride 128 elems in
// global) into LDS with +8 f16 padding per row (LDS stride 136 f16).
// D# group0/group1 per CDNA5 ISA ch.8: data_size=2B, pad every 64 DWORDs
// (=128 f16) by 4 DWORDs (=8 f16), 2-D tile -> groups 2/3 omitted (NULL).
// Issued once per calling wave; tracked by TENSORcnt (in-order per wave).
// ---------------------------------------------------------------------------
static __device__ __forceinline__ void tdm_load_tile_f16(const _Float16* gsrc,
                                                         unsigned lds_byte_off,
                                                         unsigned rows) {
    unsigned long long ga = (unsigned long long)(uintptr_t)gsrc;
    u32x4 g0;
    g0[0] = 1u;                                   // count=1 (valid), no gather
    g0[1] = lds_byte_off;                         // LDS dest byte address
    g0[2] = (unsigned)ga;                         // global_addr[31:0]
    g0[3] = (unsigned)((ga >> 32) & 0x01FFFFFFu)  // global_addr[56:32]
          | (2u << 30);                           // type=2 ("image")
    u32x8 g1;
    g1[0] = (1u << 16)                            // data_size: 2 bytes
          | (1u << 20)                            // pad_enable
          | (5u << 22)                            // pad_interval: 64 DWORDs
          | (3u << 25);                           // pad_amount: 4 DWORDs
    g1[1] = 128u << 16;                           // tensor_dim0 = 128
    g1[2] = rows << 16;                           // tensor_dim1 = rows
    g1[3] = 128u << 16;                           // tile_dim0 = 128
    g1[4] = rows;                                 // tile_dim1 = rows, tile_dim2=0
    g1[5] = 128u;                                 // tensor_dim0_stride = 128
    g1[6] = 0u;
    g1[7] = 0u;
    asm volatile("tensor_load_to_lds %0, %1" :: "s"(g0), "s"(g1) : "memory");
}

// ---------------------------------------------------------------------------
// Kernel 1: Y[M,128] = f16( X[M,1024] * W[1024,128] + bias )
// Block: 256 threads (8 waves). Tile: 128 rows x 128 cols, K-step 32.
// Register-staged pipeline (fp32 data needs f16 conversion, so no TDM here).
// ---------------------------------------------------------------------------
__global__ __launch_bounds__(256)
void proj_gemm_f16(const float* __restrict__ X, const float* __restrict__ W,
                   const float* __restrict__ bias, _Float16* __restrict__ Y) {
    __shared__ _Float16 sA[128 * 40];   // A tile row-major, padded stride 40
    __shared__ _Float16 sWt[128 * 40];  // W tile transposed: [n][k], stride 40

    const int tid  = threadIdx.x;
    const int lane = tid & 31;
    const int wave = tid >> 5;
    const int rowBase = blockIdx.x * 128;

    v8f acc[8];
#pragma unroll
    for (int t = 0; t < 8; ++t) acc[t] = vzero8();

    float4 xreg[4], wreg[4];

    auto load_tiles = [&](int k0) {
#pragma unroll
        for (int i = 0; i < 4; ++i) {
            int c4 = tid + 256 * i;              // 0..1023 float4 chunks
            {   // X tile: 128 rows x 32 cols (8 chunks/row)
                int r = c4 >> 3, c = (c4 & 7) * 4;
                xreg[i] = *(const float4*)&X[(size_t)(rowBase + r) * DIM_IN + k0 + c];
            }
            {   // W tile: 32 rows x 128 cols (32 chunks/row)
                int r = c4 >> 5, c = (c4 & 31) * 4;
                wreg[i] = *(const float4*)&W[(size_t)(k0 + r) * DIM_D + c];
            }
        }
    };
    auto store_tiles = [&]() {
#pragma unroll
        for (int i = 0; i < 4; ++i) {
            int c4 = tid + 256 * i;
            {   // X: convert to f16, 8-byte LDS store
                int r = c4 >> 3, c = (c4 & 7) * 4;
                v4h h;
                h[0] = (_Float16)xreg[i].x; h[1] = (_Float16)xreg[i].y;
                h[2] = (_Float16)xreg[i].z; h[3] = (_Float16)xreg[i].w;
                *(v4h*)&sA[r * 40 + c] = h;
            }
            {   // W: transpose scatter (4x b16 stores)
                int r = c4 >> 5, c = (c4 & 31) * 4;
                sWt[(c + 0) * 40 + r] = (_Float16)wreg[i].x;
                sWt[(c + 1) * 40 + r] = (_Float16)wreg[i].y;
                sWt[(c + 2) * 40 + r] = (_Float16)wreg[i].z;
                sWt[(c + 3) * 40 + r] = (_Float16)wreg[i].w;
            }
        }
    };

    load_tiles(0);
    for (int k0 = 0; k0 < DIM_IN; k0 += 32) {
        __syncthreads();
        store_tiles();
        __syncthreads();
        if (k0 + 32 < DIM_IN) load_tiles(k0 + 32);

        v16h a = lds_afrag(sA + wave * 16 * 40, 40, lane & 15, 0, lane);
#pragma unroll
        for (int t = 0; t < 8; ++t) {
            v16h b = lds_bfrag(sWt, 40, t * 16 + (lane & 15), 0, lane);
            acc[t] = __builtin_amdgcn_wmma_f32_16x16x32_f16(
                false, a, false, b, (short)0, acc[t], false, false);
        }
    }

    const int hf = lane >> 4;
#pragma unroll
    for (int t = 0; t < 8; ++t) {
        int col = t * 16 + (lane & 15);
        float bv = bias[col];
#pragma unroll
        for (int j = 0; j < 8; ++j) {
            int m = j + 8 * hf;   // C/D layout: vgpr j holds row j (+8 for upper half)
            Y[(size_t)(rowBase + wave * 16 + m) * DIM_D + col] = (_Float16)(acc[t][j] + bv);
        }
    }
}

// ---------------------------------------------------------------------------
// Kernel 2: flash attention. Block: 128 threads (4 waves), 64 query rows.
// Q tile + double-buffered K tiles staged by the Tensor Data Mover (wave 0),
// V tile register-staged + transposed (TDM cannot transpose). Online softmax.
// ---------------------------------------------------------------------------
__global__ __launch_bounds__(128)
void flash_attn_f16(const _Float16* __restrict__ Qf, const _Float16* __restrict__ Kf,
                    const _Float16* __restrict__ Vf, float* __restrict__ Out) {
    __shared__ _Float16 sQ[64 * 136];      // query tile, row-major, stride 136 (TDM pad)
    __shared__ _Float16 sK[2][32 * 136];   // key tile double buffer, stride 136 (TDM pad)
    __shared__ _Float16 sVt[128 * 40];     // V tile transposed [dim][key], stride 40
    __shared__ _Float16 sP[4][16 * 40];    // per-wave P tile [row][key], stride 40

    const int tid  = threadIdx.x;
    const int lane = tid & 31;
    const int wave = tid >> 5;
    const int hf   = lane >> 4;
    const int b    = blockIdx.y;
    const int qBase = blockIdx.x * 64;
    const size_t bOff = (size_t)b * SEQ * DIM_D;

    // Kick off async TDM loads: Q tile (64 rows), then K tile 0 (32 rows).
    if (wave == 0) {
        tdm_load_tile_f16(Qf + bOff + (size_t)qBase * DIM_D,
                          (unsigned)(uintptr_t)&sQ[0], 64u);
        tdm_load_tile_f16(Kf + bOff, (unsigned)(uintptr_t)&sK[0][0], 32u);
    }

    v8h vreg[4];
    auto load_v = [&](int kb) {
#pragma unroll
        for (int i = 0; i < 4; ++i) {
            int c8 = tid + 128 * i;              // 0..511 chunks of the 32x128 tile
            int r = c8 >> 4, c = (c8 & 15) * 8;
            vreg[i] = *(const v8h*)&Vf[bOff + (size_t)(kb + r) * DIM_D + c];
        }
    };
    auto store_v = [&]() {
#pragma unroll
        for (int i = 0; i < 4; ++i) {
            int c8 = tid + 128 * i;
            int r = c8 >> 4, c = (c8 & 15) * 8;
#pragma unroll
            for (int j = 0; j < 8; ++j) sVt[(c + j) * 40 + r] = vreg[i][j];
        }
    };

    float m_i[8], l_i[8];
#pragma unroll
    for (int j = 0; j < 8; ++j) { m_i[j] = -3.0e38f; l_i[j] = 0.0f; }
    v8f acc[8];
#pragma unroll
    for (int t = 0; t < 8; ++t) acc[t] = vzero8();

    const float scale = 0.08838834764831845f;  // 1/sqrt(128)

    load_v(0);
    for (int kb = 0; kb < SEQ; kb += 32) {
        const int cur = (kb >> 5) & 1;
        __syncthreads();   // everyone done with previous tiles; sVt + sK[cur^1] free
        store_v();
        if (wave == 0) {
            if (kb + 32 < SEQ) {
                // Issue next K tile into the other buffer, then wait for all
                // older TDM ops (current K, and transitively Q) to land.
                tdm_load_tile_f16(Kf + bOff + (size_t)(kb + 32) * DIM_D,
                                  (unsigned)(uintptr_t)&sK[cur ^ 1][0], 32u);
                __builtin_amdgcn_s_wait_tensorcnt(1);
            } else {
                __builtin_amdgcn_s_wait_tensorcnt(0);
            }
        }
        __syncthreads();   // publish sVt stores and TDM-delivered sQ/sK[cur]
        if (kb + 32 < SEQ) load_v(kb + 32);

        // scores S = Q_wave (16x128) * K_tile^T -> 16x32 (2 WMMA D-tiles, 4 K-steps)
        v8f sc[2];
        sc[0] = vzero8(); sc[1] = vzero8();
#pragma unroll
        for (int ks = 0; ks < 4; ++ks) {
            v16h a = lds_afrag(sQ + wave * 16 * 136, 136, lane & 15, ks * 32, lane);
#pragma unroll
            for (int t2 = 0; t2 < 2; ++t2) {
                v16h bf = lds_bfrag(&sK[cur][0], 136, t2 * 16 + (lane & 15), ks * 32, lane);
                sc[t2] = __builtin_amdgcn_wmma_f32_16x16x32_f16(
                    false, a, false, bf, (short)0, sc[t2], false, false);
            }
        }

        // online softmax: row m = j + 8*hf, spread across the 16 lanes of a half.
#pragma unroll
        for (int j = 0; j < 8; ++j) {
            float z0 = sc[0][j] * scale;
            float z1 = sc[1][j] * scale;
            float mx = fmaxf(z0, z1);
            mx = fmaxf(mx, __shfl_xor(mx, 1));
            mx = fmaxf(mx, __shfl_xor(mx, 2));
            mx = fmaxf(mx, __shfl_xor(mx, 4));
            mx = fmaxf(mx, __shfl_xor(mx, 8));
            float mnew = fmaxf(m_i[j], mx);
            float alpha = __expf(m_i[j] - mnew);
            float p0 = __expf(z0 - mnew);
            float p1 = __expf(z1 - mnew);
            float ps = p0 + p1;
            ps += __shfl_xor(ps, 1);
            ps += __shfl_xor(ps, 2);
            ps += __shfl_xor(ps, 4);
            ps += __shfl_xor(ps, 8);
            l_i[j] = l_i[j] * alpha + ps;
            m_i[j] = mnew;
            int m = j + 8 * hf;
            sP[wave][m * 40 + (lane & 15)]      = (_Float16)p0;
            sP[wave][m * 40 + 16 + (lane & 15)] = (_Float16)p1;
#pragma unroll
            for (int t = 0; t < 8; ++t) acc[t][j] *= alpha;
        }
        // sP[wave] is wave-private; LDS ops within a wave are in order.

        // O += P (16x32) * V_tile (32x128)
        v16h ap = lds_afrag(&sP[wave][0], 40, lane & 15, 0, lane);
#pragma unroll
        for (int t = 0; t < 8; ++t) {
            v16h bv = lds_bfrag(sVt, 40, t * 16 + (lane & 15), 0, lane);
            acc[t] = __builtin_amdgcn_wmma_f32_16x16x32_f16(
                false, ap, false, bv, (short)0, acc[t], false, false);
        }
    }

    // epilogue: normalize and store fp32
    float rl[8];
#pragma unroll
    for (int j = 0; j < 8; ++j) rl[j] = 1.0f / l_i[j];
#pragma unroll
    for (int t = 0; t < 8; ++t) {
        int col = t * 16 + (lane & 15);
#pragma unroll
        for (int j = 0; j < 8; ++j) {
            int m = j + 8 * hf;
            Out[bOff + (size_t)(qBase + wave * 16 + m) * DIM_D + col] = acc[t][j] * rl[j];
        }
    }
}

extern "C" void kernel_launch(void* const* d_in, const int* in_sizes, int n_in,
                              void* d_out, int out_size, void* d_ws, size_t ws_size,
                              hipStream_t stream) {
    const float* query = (const float*)d_in[0];
    const float* key   = (const float*)d_in[1];
    const float* value = (const float*)d_in[2];
    const float* Wq = (const float*)d_in[3];
    const float* bq = (const float*)d_in[4];
    const float* Wk = (const float*)d_in[5];
    const float* bk = (const float*)d_in[6];
    const float* Wv = (const float*)d_in[7];
    const float* bv = (const float*)d_in[8];
    float* out = (float*)d_out;

    _Float16* qf = (_Float16*)d_ws;
    _Float16* kf = qf + (size_t)MROWS * DIM_D;
    _Float16* vf = kf + (size_t)MROWS * DIM_D;

    dim3 g1(MROWS / 128), b1(256);
    proj_gemm_f16<<<g1, b1, 0, stream>>>(query, Wq, bq, qf);
    proj_gemm_f16<<<g1, b1, 0, stream>>>(key,   Wk, bk, kf);
    proj_gemm_f16<<<g1, b1, 0, stream>>>(value, Wv, bv, vf);

    dim3 g2(SEQ / 64, BATCH), b2(128);
    flash_attn_f16<<<g2, b2, 0, stream>>>(qf, kf, vf, out);
}
